// GPT_79001628442981
// MI455X (gfx1250) — matrix operand order repaired and checked by reading, not compiled, not run on hardware
//
#include <hip/hip_runtime.h>
#include <cstdint>

typedef __attribute__((ext_vector_type(16))) _Float16 v16h;
typedef __attribute__((ext_vector_type(8)))  float    v8f;

#define Bc   4
#define Tc   1024
#define Vc   32000
#define Dc   1024
#define Lc   8
#define Hc   16
#define HDc  64
#define Mrows (Bc*Tc)

union F16Frag { v16h v; uint4 q[2]; };

static __device__ __forceinline__ v8f wmma16(const F16Frag& a, const F16Frag& b, v8f c) {
  // D = A(16x32 f16) * B(32x16 f16) + C(16x16 f32)
  return __builtin_amdgcn_wmma_f32_16x16x32_f16(false, a.v, false, b.v, (short)0, c, false, false);
}

// CDNA5 async copy: 16B per lane, global -> LDS, tracked by ASYNCcnt.
static __device__ __forceinline__ void async_load_b128(unsigned ldsOff, unsigned long long gaddr) {
  asm volatile("global_load_async_to_lds_b128 %0, %1, off"
               :: "v"(ldsOff), "v"(gaddr) : "memory");
}
static __device__ __forceinline__ void wait_async0() {
  asm volatile("s_wait_asynccnt 0" ::: "memory");
}

// ---------------------------------------------------------------- embedding
__global__ __launch_bounds__(256) void k_embed(const int* __restrict__ idx,
                                               const float* __restrict__ tok,
                                               const float* __restrict__ pos,
                                               float* __restrict__ x) {
  int row = blockIdx.x;            // b*T + t
  int t   = row & (Tc - 1);
  int tk  = idx[row];
  int i   = threadIdx.x * 4;
  float4 a = *(const float4*)&tok[(size_t)tk * Dc + i];
  float4 p = *(const float4*)&pos[(size_t)t  * Dc + i];
  float4 r; r.x = a.x + p.x; r.y = a.y + p.y; r.z = a.z + p.z; r.w = a.w + p.w;
  *(float4*)&x[(size_t)row * Dc + i] = r;
}

// ---------------------------------------------------------------- layernorm -> f16
__global__ __launch_bounds__(256) void k_ln(const float* __restrict__ x,
                                            const float* __restrict__ g,
                                            const float* __restrict__ bta,
                                            _Float16* __restrict__ out) {
  __shared__ float rs[256], rq[256];
  int row = blockIdx.x, tid = threadIdx.x;
  const float* xr = x + (size_t)row * Dc;
  float4 v = *(const float4*)&xr[tid * 4];
  rs[tid] = v.x + v.y + v.z + v.w;
  rq[tid] = v.x*v.x + v.y*v.y + v.z*v.z + v.w*v.w;
  __syncthreads();
  for (int st = 128; st > 0; st >>= 1) {
    if (tid < st) { rs[tid] += rs[tid + st]; rq[tid] += rq[tid + st]; }
    __syncthreads();
  }
  float mean = rs[0] * (1.0f / Dc);
  float var  = rq[0] * (1.0f / Dc) - mean * mean;
  float rstd = rsqrtf(var + 1e-5f);
  int i = tid * 4;
  _Float16* o = out + (size_t)row * Dc + i;
  o[0] = (_Float16)((v.x - mean) * rstd * g[i+0] + bta[i+0]);
  o[1] = (_Float16)((v.y - mean) * rstd * g[i+1] + bta[i+1]);
  o[2] = (_Float16)((v.z - mean) * rstd * g[i+2] + bta[i+2]);
  o[3] = (_Float16)((v.w - mean) * rstd * g[i+3] + bta[i+3]);
}

// ---------------------------------------------------------------- WMMA GEMM
// C[M,N] = A[M,K](f16) * B[K,N](fp32 weights, converted to f16 while staging)
// MODE 0: outH = f16(acc+bias [gelu])       (row-major, ld = N)
// MODE 1: outF = [resid +] acc [+ bias]     (f32, row-major, ld = N)
// MODE 2: per-head transposed f16 store: outH[((b*H+h)*HD+d)*T + t]
// Block tile 128x128, 8 waves, each wave 32(M) x 64(N) = 8 wmma tiles.
template<int MODE, bool HAS_BIAS, bool GELU, bool HAS_RESID>
__global__ __launch_bounds__(256) void k_gemm(const _Float16* __restrict__ A, int lda,
                                              const float* __restrict__ Bw, int N, int K,
                                              const float* __restrict__ bias,
                                              const float* resid,
                                              float* outF,
                                              _Float16* __restrict__ outH) {
  __shared__ _Float16 sA[128][32];    // A tile, row-major
  __shared__ _Float16 sBt[128][32];   // B tile, transposed [n][k]
  int tid  = threadIdx.x;
  int n0   = blockIdx.x * 128;
  int m0   = blockIdx.y * 128;
  int lane = tid & 31, wid = tid >> 5;
  int wm = wid >> 1, wn = wid & 1;    // 4 (M) x 2 (N) waves, 32x64 each

  v8f acc[2][4] = {};

  int kbA = (lane & 16) ? 8 : 0;      // A-frag K base
  int kbB = (lane & 16) ? 8 : 0;      // B-frag K base

  for (int k0 = 0; k0 < K; k0 += 32) {
    __syncthreads();
    // ---- stage A via CDNA5 async-to-LDS (2 x 16B per thread)
#pragma unroll
    for (int cc = 0; cc < 2; ++cc) {
      int c   = tid + cc * 256;       // 0..511
      int row = c >> 2;               // 0..127
      int c8  = (c & 3) * 8;          // 0/8/16/24
      unsigned lo = (unsigned)(uintptr_t)&sA[row][c8];
      unsigned long long ga =
          (unsigned long long)(uintptr_t)(A + (size_t)(m0 + row) * lda + k0 + c8);
      async_load_b128(lo, ga);
    }
    // ---- stage B: read fp32, convert to f16, store transposed
#pragma unroll
    for (int cc = 0; cc < 2; ++cc) {
      int c  = tid + cc * 256;
      int kk = c >> 4;                // 0..31
      int n8 = (c & 15) * 8;          // 0..120
      const float* bp = Bw + (size_t)(k0 + kk) * N + n0 + n8;
      float4 f0 = *(const float4*)bp;
      float4 f1 = *(const float4*)(bp + 4);
      if (k0 + 32 < K) __builtin_prefetch(bp + (size_t)32 * N, 0, 1);
      sBt[n8+0][kk] = (_Float16)f0.x; sBt[n8+1][kk] = (_Float16)f0.y;
      sBt[n8+2][kk] = (_Float16)f0.z; sBt[n8+3][kk] = (_Float16)f0.w;
      sBt[n8+4][kk] = (_Float16)f1.x; sBt[n8+5][kk] = (_Float16)f1.y;
      sBt[n8+6][kk] = (_Float16)f1.z; sBt[n8+7][kk] = (_Float16)f1.w;
    }
    wait_async0();
    __syncthreads();
    // ---- compute: A fragments (2 M-subtiles), reuse B fragment across both
    F16Frag fa[2];
#pragma unroll
    for (int mi = 0; mi < 2; ++mi) {
      int r = wm * 32 + mi * 16 + (lane & 15);
      fa[mi].q[0] = *(const uint4*)&sA[r][kbA];
      fa[mi].q[1] = *(const uint4*)&sA[r][kbA + 16];
    }
#pragma unroll
    for (int j = 0; j < 4; ++j) {
      F16Frag fb;
      int n = wn * 64 + j * 16 + (lane & 15);
      fb.q[0] = *(const uint4*)&sBt[n][kbB];
      fb.q[1] = *(const uint4*)&sBt[n][kbB + 16];
#pragma unroll
      for (int mi = 0; mi < 2; ++mi)
        acc[mi][j] = wmma16(fa[mi], fb, acc[mi][j]);
    }
  }

  int rbase = (lane & 16) ? 8 : 0;
#pragma unroll
  for (int mi = 0; mi < 2; ++mi) {
#pragma unroll
    for (int j = 0; j < 4; ++j) {
      int nn = n0 + wn * 64 + j * 16 + (lane & 15);
      float bnn = HAS_BIAS ? bias[nn] : 0.0f;
#pragma unroll
      for (int r = 0; r < 8; ++r) {
        int mm = m0 + wm * 32 + mi * 16 + r + rbase;
        float v = acc[mi][j][r] + bnn;
        if constexpr (GELU) v = 0.5f * v * (1.0f + erff(v * 0.70710678118654752f));
        size_t oi = (size_t)mm * N + nn;
        if constexpr (MODE == 0) {
          outH[oi] = (_Float16)v;
        } else if constexpr (MODE == 1) {
          float rv = HAS_RESID ? resid[oi] : 0.0f;
          outF[oi] = rv + v;
        } else {                      // transposed per-head V store (N==Dc)
          int bb = mm >> 10;          // / Tc
          int tt = mm & (Tc - 1);
          int hh = nn >> 6;           // / HDc
          int dd = nn & (HDc - 1);
          outH[(size_t)((bb * Hc + hh) * HDc + dd) * Tc + tt] = (_Float16)v;
        }
      }
    }
  }
}

// ---------------------------------------------------------------- attention
// one workgroup per (b, h, 16-row query tile); dynamic LDS: S f32 64KB + P f16 32KB
__global__ __launch_bounds__(256) void k_attn(const _Float16* __restrict__ q16,
                                              const _Float16* __restrict__ k16,
                                              const _Float16* __restrict__ vT16,
                                              _Float16* __restrict__ o16) {
  extern __shared__ char smem[];
  float*    S = (float*)smem;                                   // [16][1024]
  _Float16* P = (_Float16*)(smem + 16 * 1024 * sizeof(float));  // [16][1024]
  __shared__ float part[16][16], rmax[16], rsum[16];

  int qt = blockIdx.x, h = blockIdx.y, b = blockIdx.z;
  int tid = threadIdx.x, lane = tid & 31, wid = tid >> 5;
  const float scale = 0.125f;  // 1/sqrt(HD)

  int kb2   = (lane & 16) ? 8 : 0;
  int rbase = (lane & 16) ? 8 : 0;

  // ---- S = Q K^T * scale : each of 8 waves does 8 score tiles
  F16Frag fq[2];
  {
    int qrow = qt * 16 + (lane & 15);
    int kb   = (lane & 16) ? 8 : 0;
    const _Float16* qp = q16 + ((size_t)(b * Tc + qrow)) * Dc + h * HDc;
#pragma unroll
    for (int c = 0; c < 2; ++c) {
      fq[c].q[0] = *(const uint4*)(qp + c * 32 + kb);
      fq[c].q[1] = *(const uint4*)(qp + c * 32 + kb + 16);
    }
  }
#pragma unroll
  for (int j = 0; j < 8; ++j) {
    int n0 = (wid * 8 + j) * 16;
    v8f acc = {};
    int krow = n0 + (lane & 15);
    const _Float16* kp = k16 + ((size_t)(b * Tc + krow)) * Dc + h * HDc;
#pragma unroll
    for (int c = 0; c < 2; ++c) {
      F16Frag fk;   // B[k][n] = K[n0+n][k] : contiguous along k
      fk.q[0] = *(const uint4*)(kp + c * 32 + kb2);
      fk.q[1] = *(const uint4*)(kp + c * 32 + kb2 + 16);
      acc = wmma16(fq[c], fk, acc);
    }
#pragma unroll
    for (int r = 0; r < 8; ++r)
      S[(r + rbase) * 1024 + n0 + (lane & 15)] = acc[r] * scale;
  }
  __syncthreads();

  // ---- causal softmax (16 threads per row)
  {
    int m = tid >> 4, s = tid & 15;
    int qg = qt * 16 + m;
    float mx = -3.0e38f;
    for (int c = s; c < 1024; c += 16)
      if (c <= qg) mx = fmaxf(mx, S[m * 1024 + c]);
    part[m][s] = mx;
    __syncthreads();
    if (s == 0) {
      float v = part[m][0];
      for (int i = 1; i < 16; ++i) v = fmaxf(v, part[m][i]);
      rmax[m] = v;
    }
    __syncthreads();
    float rm = rmax[m], sum = 0.0f;
    for (int c = s; c < 1024; c += 16) {
      float e = (c <= qg) ? expf(S[m * 1024 + c] - rm) : 0.0f;
      S[m * 1024 + c] = e;
      sum += e;
    }
    part[m][s] = sum;
    __syncthreads();
    if (s == 0) {
      float v = 0.0f;
      for (int i = 0; i < 16; ++i) v += part[m][i];
      rsum[m] = v;
    }
    __syncthreads();
    float inv = 1.0f / rsum[m];
    for (int c = s; c < 1024; c += 16)
      P[m * 1024 + c] = (_Float16)(S[m * 1024 + c] * inv);
  }
  __syncthreads();

  // ---- O = P @ V : 4 waves, one 16x16 n-tile each, K=1024 in 32-chunks
  if (wid < 4) {
    int nt = wid;
    v8f acc = {};
    int prow = lane & 15;
    int kbA  = (lane & 16) ? 8 : 0;
    const _Float16* vp = vT16 + (size_t)((b * Hc + h) * HDc + nt * 16 + (lane & 15)) * Tc;
    for (int kc = 0; kc < 1024; kc += 32) {
      F16Frag fp, fv;
      const _Float16* pp = P + prow * 1024 + kc + kbA;
      fp.q[0] = *(const uint4*)pp;
      fp.q[1] = *(const uint4*)(pp + 16);
      const _Float16* vv = vp + kc + kb2;   // V^T: contiguous along k
      fv.q[0] = *(const uint4*)vv;
      fv.q[1] = *(const uint4*)(vv + 16);
      acc = wmma16(fp, fv, acc);
    }
#pragma unroll
    for (int r = 0; r < 8; ++r) {
      int t   = qt * 16 + r + rbase;
      int col = h * HDc + nt * 16 + (lane & 15);
      o16[((size_t)(b * Tc + t)) * Dc + col] = (_Float16)acc[r];
    }
  }
}

// ---------------------------------------------------------------- host
template<int MODE, bool HB, bool GELU, bool HR>
static void launch_gemm(const _Float16* A, int lda, const float* Bw, int N, int K,
                        const float* bias, const float* resid, float* oF, _Float16* oH,
                        hipStream_t s) {
  dim3 grid(N / 128, Mrows / 128);
  k_gemm<MODE, HB, GELU, HR><<<grid, 256, 0, s>>>(A, lda, Bw, N, K, bias, resid, oF, oH);
}

extern "C" void kernel_launch(void* const* d_in, const int* in_sizes, int n_in,
                              void* d_out, int out_size, void* d_ws, size_t ws_size,
                              hipStream_t stream) {
  (void)in_sizes; (void)n_in; (void)out_size; (void)ws_size;
  const int*   idx     = (const int*)d_in[0];
  const float* tok_emb = (const float*)d_in[1];
  const float* pos_emb = (const float*)d_in[2];
  const float* ln1_g   = (const float*)d_in[3];
  const float* ln1_b   = (const float*)d_in[4];
  const float* Wq      = (const float*)d_in[5];
  const float* bq      = (const float*)d_in[6];
  const float* Wk      = (const float*)d_in[7];
  const float* bk      = (const float*)d_in[8];
  const float* Wv      = (const float*)d_in[9];
  const float* bv      = (const float*)d_in[10];
  const float* Wo      = (const float*)d_in[11];
  const float* bo      = (const float*)d_in[12];
  const float* ln2_g   = (const float*)d_in[13];
  const float* ln2_b   = (const float*)d_in[14];
  const float* W1      = (const float*)d_in[15];
  const float* b1      = (const float*)d_in[16];
  const float* W2      = (const float*)d_in[17];
  const float* b2      = (const float*)d_in[18];
  const float* lnf_g   = (const float*)d_in[19];
  const float* lnf_b   = (const float*)d_in[20];
  const float* head_W  = (const float*)d_in[21];
  float* out = (float*)d_out;

  char* w = (char*)d_ws;
  float*    x    = (float*)w;    w += (size_t)Mrows * Dc * 4;      // residual stream
  _Float16* h16  = (_Float16*)w; w += (size_t)Mrows * Dc * 2;      // LN output
  _Float16* q16  = (_Float16*)w; w += (size_t)Mrows * Dc * 2;
  _Float16* k16s = (_Float16*)w; w += (size_t)Mrows * Dc * 2;
  _Float16* vT16 = (_Float16*)w; w += (size_t)Mrows * Dc * 2;      // [B,H,HD,T]
  _Float16* o16  = (_Float16*)w; w += (size_t)Mrows * Dc * 2;
  _Float16* m16  = (_Float16*)w; w += (size_t)Mrows * 4 * Dc * 2;  // MLP mid

  (void)hipFuncSetAttribute(reinterpret_cast<const void*>(k_attn),
                            hipFuncAttributeMaxDynamicSharedMemorySize, 98304);

  k_embed<<<Mrows, 256, 0, stream>>>(idx, tok_emb, pos_emb, x);

  for (int l = 0; l < Lc; ++l) {
    size_t wOff = (size_t)l * Dc * Dc;
    size_t bOff = (size_t)l * Dc;
    k_ln<<<Mrows, 256, 0, stream>>>(x, ln1_g + bOff, ln1_b + bOff, h16);
    launch_gemm<0, true, false, false>(h16, Dc, Wq + wOff, Dc, Dc, bq + bOff,
                                       nullptr, nullptr, q16, stream);
    launch_gemm<0, true, false, false>(h16, Dc, Wk + wOff, Dc, Dc, bk + bOff,
                                       nullptr, nullptr, k16s, stream);
    launch_gemm<2, true, false, false>(h16, Dc, Wv + wOff, Dc, Dc, bv + bOff,
                                       nullptr, nullptr, vT16, stream);
    dim3 ag(Tc / 16, Hc, Bc);
    k_attn<<<ag, 256, 98304, stream>>>(q16, k16s, vT16, o16);
    launch_gemm<1, true, false, true>(o16, Dc, Wo + wOff, Dc, Dc, bo + bOff,
                                      x, x, nullptr, stream);
    k_ln<<<Mrows, 256, 0, stream>>>(x, ln2_g + bOff, ln2_b + bOff, h16);
    launch_gemm<0, true, true, false>(h16, Dc, W1 + (size_t)l * Dc * 4 * Dc, 4 * Dc, Dc,
                                      b1 + (size_t)l * 4 * Dc, nullptr, nullptr, m16, stream);
    launch_gemm<1, true, false, true>(m16, 4 * Dc, W2 + (size_t)l * 4 * Dc * Dc, Dc, 4 * Dc,
                                      b2 + bOff, x, x, nullptr, stream);
  }
  k_ln<<<Mrows, 256, 0, stream>>>(x, lnf_g, lnf_b, h16);
  launch_gemm<1, false, false, false>(h16, Dc, head_W, Vc, Dc, nullptr,
                                      nullptr, out, nullptr, stream);
}
